// POSTagger_8254927143514
// MI455X (gfx1250) — matrix operand order repaired and checked by reading, not compiled, not run on hardware
//
#include <hip/hip_runtime.h>
#include <hip/hip_bf16.h>

// ---------------------------------------------------------------------------
// POS tagger: char-CNN + word-emb concat -> pregate GEMM (WMMA bf16)
//             -> sequential LSTM (1 WGP, LDS state, float4 GEMV)
//             -> dense + log_softmax
// Sizes (from reference): T=4096 words, CE=10, NF=32, WE=250, IN=282, HID=250
// ---------------------------------------------------------------------------

typedef unsigned int  uint;
typedef unsigned short ushort;
typedef __attribute__((ext_vector_type(16))) __bf16 v16bf;
typedef __attribute__((ext_vector_type(8)))  float  v8f;

#define T_WORDS   4096
#define MAX_CHARS 16
#define CE_DIM    10
#define NF        32
#define WE        250
#define IN_DIM    282      // NF + WE
#define K_PAD     288      // 9 * 32
#define KCHUNKS   9
#define HID       250
#define NGATE     1000     // 4 * HID
#define N_PAD     1024     // 64 tiles of 16
#define NTILES    64
#define K4CHUNKS  64       // ceil(250/4) padded to 64 (K padded to 256)
#define TAGS      50

// ---- helpers ---------------------------------------------------------------
__device__ __forceinline__ ushort f2bf(float x) {
    uint u = __builtin_bit_cast(uint, x);
    u = (u + 0x7FFFu + ((u >> 16) & 1u)) >> 16;   // round-to-nearest-even
    return (ushort)u;
}
__device__ __forceinline__ float sigmoidf(float x) {
    return 1.0f / (1.0f + __expf(-x));
}

// ---- workspace layout (bytes, all 16B-aligned) ------------------------------
#define WS_XBF    0u            // ushort [4096*288]      2,359,296 B
#define WS_WB     2359296u      // uint   [9*64*32*8]       589,824 B
#define WS_PRE    2949120u      // float  [4096*1024]    16,777,216 B
#define WS_WHH    19726336u     // float4 [64*1024]       1,048,576 B
#define WS_BSUM   20774912u     // float  [1024]              4,096 B
#define WS_HS     20779008u     // float  [4096*256]      4,194,304 B
// total ~25 MB

// ===========================================================================
// 1) Pre-swizzle W_ih into WMMA B-fragment order, bf16, zero-padded.
//    B (32x16 bf16) lane layout: lane L -> col n = L&15, koff = (L>>4)*8,
//    VGPR j holds K pair (16*(j>>2)+koff+2*(j&3), +1).
//    Fragment buffer: wB[((kc*64 + ntile)*32 + lane)*8 + j]  (packed bf16x2)
// ===========================================================================
__global__ void pt_pack_wih(const float* __restrict__ w_ih, uint* __restrict__ wB) {
    int idx = blockIdx.x * blockDim.x + threadIdx.x;   // 147456 total
    int j    = idx & 7;
    int lane = (idx >> 3) & 31;
    int nt   = (idx >> 8) & 63;
    int kc   = idx >> 14;
    int n    = nt * 16 + (lane & 15);
    int K    = kc * 32 + ((j >> 2) * 16) + ((lane >> 4) * 8) + 2 * (j & 3);
    ushort lo = (n < NGATE && K     < IN_DIM) ? f2bf(w_ih[n * IN_DIM + K])     : (ushort)0;
    ushort hi = (n < NGATE && K + 1 < IN_DIM) ? f2bf(w_ih[n * IN_DIM + K + 1]) : (ushort)0;
    wB[idx] = (uint)lo | ((uint)hi << 16);
}

// W_hh [1000,250] -> f32 float4-along-K transposed layout:
//   whh_q[(k4*1024 + j)] = { W[j][4k4], W[j][4k4+1], W[j][4k4+2], W[j][4k4+3] }
// zero-padded in both j (1000->1024) and k (250->256).
__global__ void pt_pack_whh4(const float* __restrict__ w_hh, float* __restrict__ whh_q) {
    int idx = blockIdx.x * blockDim.x + threadIdx.x;   // 262144 total floats
    int comp = idx & 3;
    int j    = (idx >> 2) & 1023;
    int k4   = idx >> 12;
    int k    = 4 * k4 + comp;
    whh_q[idx] = (k < HID && j < NGATE) ? w_hh[j * HID + k] : 0.0f;
}

__global__ void pt_pack_bias(const float* __restrict__ b_ih, const float* __restrict__ b_hh,
                             float* __restrict__ bsum) {
    int j = blockIdx.x * blockDim.x + threadIdx.x;     // 1024 total
    bsum[j] = (j < NGATE) ? (b_ih[j] + b_hh[j]) : 0.0f;
}

// ===========================================================================
// 2) char-CNN + embedding concat -> X bf16 [4096, 288]  (one word per block)
// ===========================================================================
__global__ void pt_build_x(const int* __restrict__ char_idx, const int* __restrict__ word_idx,
                           const float* __restrict__ char_emb, const float* __restrict__ word_emb,
                           const float* __restrict__ conv_w, const float* __restrict__ conv_b,
                           ushort* __restrict__ xbf) {
    const int w = blockIdx.x, tid = threadIdx.x;       // blockDim = 64
    __shared__ float ce[MAX_CHARS][CE_DIM];
    for (int i = tid; i < MAX_CHARS * CE_DIM; i += 64) {
        int p = i / CE_DIM, c = i % CE_DIM;
        ce[p][c] = char_emb[char_idx[w * MAX_CHARS + p] * CE_DIM + c];
    }
    __syncthreads();

    if (tid < NF) {
        const int f = tid;
        float wr[CE_DIM * 3];
        #pragma unroll
        for (int i = 0; i < CE_DIM * 3; ++i) wr[i] = conv_w[f * CE_DIM * 3 + i];
        const float b = conv_b[f];
        float mx = -3.4e38f;
        #pragma unroll
        for (int p = 0; p < MAX_CHARS; ++p) {
            float s = b;
            #pragma unroll
            for (int kk = 0; kk < 3; ++kk) {
                int q = p + kk - 1;
                if (q >= 0 && q < MAX_CHARS)
                    #pragma unroll
                    for (int c = 0; c < CE_DIM; ++c) s += ce[q][c] * wr[c * 3 + kk];
            }
            mx = fmaxf(mx, s);
        }
        xbf[w * K_PAD + f] = f2bf(mx);
    }
    const int wi = word_idx[w];
    for (int k = tid; k < WE; k += 64)
        xbf[w * K_PAD + NF + k] = f2bf(word_emb[wi * WE + k]);
    if (tid < K_PAD - IN_DIM) xbf[w * K_PAD + IN_DIM + tid] = 0;   // pad 282..287
}

// ===========================================================================
// 3) pregate = X @ W_ih^T + (b_ih + b_hh)  via v_wmma_f32_16x16x32_bf16
//    grid(256 Mtiles, 16) x block(128) : one wave = one 16x16 tile,
//    wave's ntile = blockIdx.y*4 + waveId; 9 K-chunks of 32.
// ===========================================================================
__global__ void pt_gemm_pregate(const ushort* __restrict__ xbf, const uint* __restrict__ wB,
                                const float* __restrict__ bsum, float* __restrict__ pregate) {
    const int lane  = threadIdx.x & 31;
    const int wv    = threadIdx.x >> 5;
    const int mtile = blockIdx.x;
    const int ntile = blockIdx.y * 4 + wv;

    const ushort* arow = xbf + (mtile * 16 + (lane & 15)) * K_PAD;
    const int kb = (lane >> 4) * 8;                      // A lane K sub-offset
    v8f c = {};

    #pragma unroll
    for (int kc = 0; kc < KCHUNKS; ++kc) {
        union { uint u[8]; v16bf v; } A, B;
        #pragma unroll
        for (int j = 0; j < 4; ++j)
            A.u[j] = *(const uint*)(arow + kc * 32 + kb + 2 * j);
        #pragma unroll
        for (int j = 4; j < 8; ++j)
            A.u[j] = *(const uint*)(arow + kc * 32 + 16 + kb + 2 * (j - 4));
        const uint* bp = wB + ((kc * NTILES + ntile) * 32 + lane) * 8;
        #pragma unroll
        for (int j = 0; j < 8; ++j) B.u[j] = bp[j];
        c = __builtin_amdgcn_wmma_f32_16x16x32_bf16(false, A.v, false, B.v,
                                                    (short)0, c, false, false);
    }

    const int n     = ntile * 16 + (lane & 15);
    const float bb  = bsum[n];
    const int mrow0 = mtile * 16 + (lane >> 4) * 8;      // C layout: VGPR r -> M = r + 8*(lane>=16)
    #pragma unroll
    for (int r = 0; r < 8; ++r)
        pregate[(mrow0 + r) * N_PAD + n] = c[r] + bb;
}

// ===========================================================================
// 4) Sequential LSTM: single 1024-thread workgroup (32 waves, one WGP).
//    thread j owns gate column j. Recurrent GEMV vectorized by 4 along K:
//    per 4 MACs = 1 global_load_b128 (W, coalesced, L2-resident)
//               + 1 ds_load_b128 (h broadcast) + 4 v_fma.
//    Two barriers per step; c state lives in registers of threads 0..249.
// ===========================================================================
__global__ void pt_lstm(const float* __restrict__ pregate, const float4* __restrict__ whh_q,
                        float* __restrict__ hs) {
    const int j = threadIdx.x;                 // 0..1023
    __shared__ float4 h4[K4CHUNKS];            // h, zero-padded to 256
    __shared__ float  gbuf[1024];
    float* h = (float*)h4;
    if (j < 256) h[j] = 0.0f;
    float c_reg = 0.0f;
    __syncthreads();

    const float4* wp = whh_q + j;              // column j, stride 1024 float4s per k4

    for (int t = 0; t < T_WORDS; ++t) {
        if (t + 1 < T_WORDS)
            __builtin_prefetch(&pregate[(t + 1) * N_PAD + j], 0, 1);
        float acc = pregate[t * N_PAD + j];    // padded cols computed as 0
        #pragma unroll 8
        for (int k4 = 0; k4 < K4CHUNKS; ++k4) {
            float4 w4 = wp[k4 * 1024];
            float4 hv = h4[k4];
            acc = fmaf(w4.x, hv.x, acc);
            acc = fmaf(w4.y, hv.y, acc);
            acc = fmaf(w4.z, hv.z, acc);
            acc = fmaf(w4.w, hv.w, acc);
        }
        gbuf[j] = acc;
        __syncthreads();                       // all h reads + gbuf writes of step t done

        if (j < HID) {
            float ig = sigmoidf(gbuf[j]);
            float fg = sigmoidf(gbuf[HID + j]);
            float gg = tanhf(gbuf[2 * HID + j]);
            float og = sigmoidf(gbuf[3 * HID + j]);
            c_reg = fg * c_reg + ig * gg;
            float hn = og * tanhf(c_reg);
            hs[t * 256 + j] = hn;
            h[j] = hn;                         // safe: step-t h reads finished above
        }
        __syncthreads();                       // publish h, free gbuf for step t+1
    }
}

// ===========================================================================
// 5) dense + log_softmax : one word per 64-thread block
// ===========================================================================
__global__ void pt_dense_lsm(const float* __restrict__ hs, const float* __restrict__ dense_w,
                             const float* __restrict__ dense_b, float* __restrict__ out) {
    const int t = blockIdx.x, g = threadIdx.x;   // blockDim = 64
    __shared__ float lg[TAGS];
    if (g < TAGS) {
        float acc = dense_b[g];
        const float* hr = hs + t * 256;
        #pragma unroll 10
        for (int k = 0; k < HID; ++k) acc += hr[k] * dense_w[g * HID + k];
        lg[g] = acc;
    }
    __syncthreads();
    float m = -3.4e38f;
    #pragma unroll
    for (int i = 0; i < TAGS; ++i) m = fmaxf(m, lg[i]);
    float s = 0.0f;
    #pragma unroll
    for (int i = 0; i < TAGS; ++i) s += __expf(lg[i] - m);
    if (g < TAGS) out[t * TAGS + g] = lg[g] - m - __logf(s);
}

// ===========================================================================
extern "C" void kernel_launch(void* const* d_in, const int* in_sizes, int n_in,
                              void* d_out, int out_size, void* d_ws, size_t ws_size,
                              hipStream_t stream) {
    const int*   char_idx = (const int*)  d_in[0];
    const int*   word_idx = (const int*)  d_in[1];
    const float* char_emb = (const float*)d_in[2];
    const float* word_emb = (const float*)d_in[3];
    const float* conv_w   = (const float*)d_in[4];
    const float* conv_b   = (const float*)d_in[5];
    const float* w_ih     = (const float*)d_in[6];
    const float* w_hh     = (const float*)d_in[7];
    const float* b_ih     = (const float*)d_in[8];
    const float* b_hh     = (const float*)d_in[9];
    const float* dense_w  = (const float*)d_in[10];
    const float* dense_b  = (const float*)d_in[11];
    float* out = (float*)d_out;

    char* ws = (char*)d_ws;
    ushort* xbf    = (ushort*)(ws + WS_XBF);
    uint*   wB     = (uint*)  (ws + WS_WB);
    float*  pre    = (float*) (ws + WS_PRE);
    float*  whh_q  = (float*) (ws + WS_WHH);
    float*  bsum   = (float*) (ws + WS_BSUM);
    float*  hs     = (float*) (ws + WS_HS);

    pt_pack_wih  <<<576,  256, 0, stream>>>(w_ih, wB);
    pt_pack_whh4 <<<1024, 256, 0, stream>>>(w_hh, whh_q);
    pt_pack_bias <<<4,    256, 0, stream>>>(b_ih, b_hh, bsum);
    pt_build_x   <<<T_WORDS, 64, 0, stream>>>(char_idx, word_idx, char_emb, word_emb,
                                              conv_w, conv_b, xbf);
    pt_gemm_pregate<<<dim3(T_WORDS / 16, NTILES / 4), 128, 0, stream>>>(xbf, wB, bsum, pre);
    pt_lstm      <<<1, 1024, 0, stream>>>(pre, (const float4*)whh_q, hs);
    pt_dense_lsm <<<T_WORDS, 64, 0, stream>>>(hs, dense_w, dense_b, out);
}
